// ResidualHVQ_64570538328100
// MI455X (gfx1250) — matrix operand order repaired and checked by reading, not compiled, not run on hardware
//
#include <hip/hip_runtime.h>
#include <math.h>

typedef __attribute__((ext_vector_type(16))) _Float16 v16h;
typedef __attribute__((ext_vector_type(8)))  _Float16 v8h;
typedef __attribute__((ext_vector_type(8)))  float    v8f;

#define BB 8
#define NN 2048
#define HH 8
#define DD 64
#define MM 1024
#define RR 4
#define FF 512
#define TT (BB * NN)            // 16384 tokens
#define TOK_PER_BLOCK 256
#define NTHREADS 512            // 16 waves

#define LDS_CB_BYTES  (MM * DD * 2)               // 131072: f16 normalized codebook (one head)
#define LDS_RES_BYTES (TOK_PER_BLOCK * DD * 4)    // 65536: f32 residual tile
#define LDS_IDX_BYTES (16 * 16 * 4)               // 1024: per-wave winning indices
#define LDS_TOTAL (LDS_CB_BYTES + LDS_RES_BYTES + LDS_IDX_BYTES)

// ---------------------------------------------------------------------------
// Kernel 0: L2-normalize codebooks, store as f16 row-major [h][m][d] in ws.
// ---------------------------------------------------------------------------
__global__ void cb_norm_kernel(const float* __restrict__ cb,
                               _Float16* __restrict__ cn) {
  const int w = threadIdx.x >> 5;
  const int lane = threadIdx.x & 31;
  const int row = blockIdx.x * 8 + w;              // 0 .. H*M-1
  const float* src = cb + (size_t)row * DD + lane * 2;
  float a = src[0];
  float b = src[1];
  float s = a * a + b * b;
#pragma unroll
  for (int off = 16; off >= 1; off >>= 1) s += __shfl_xor(s, off, 32);
  const float inv = 1.0f / fmaxf(sqrtf(s), 1e-12f);
  _Float16* dst = cn + (size_t)row * DD + lane * 2;
  dst[0] = (_Float16)(a * inv);
  dst[1] = (_Float16)(b * inv);
}

__global__ void zero_kernel(int* __restrict__ p, int n) {
  int i = blockIdx.x * blockDim.x + threadIdx.x;
  if (i < n) p[i] = 0;
}

// Load one 16-codeword B tile (K=0..63 -> two 16x16x32 fragments).
// Contiguous K order used here (A matches): lanes 0-15 hold K=0..15, lanes 16-31 hold K=16..31,
// VGPR v packs rows (K) 2v,2v+1 -> 16 contiguous f16 per lane per fragment.
__device__ __forceinline__ void loadB(const _Float16* p, v16h& o0, v16h& o1) {
  v8h l0 = *(const v8h*)(p + 0);
  v8h h0 = *(const v8h*)(p + 8);
  v8h l1 = *(const v8h*)(p + 32);
  v8h h1 = *(const v8h*)(p + 40);
  o0 = __builtin_shufflevector(l0, h0, 0, 1, 2, 3, 4, 5, 6, 7,
                               8, 9, 10, 11, 12, 13, 14, 15);
  o1 = __builtin_shufflevector(l1, h1, 0, 1, 2, 3, 4, 5, 6, 7,
                               8, 9, 10, 11, 12, 13, 14, 15);
}

// ---------------------------------------------------------------------------
// Kernel 1: main residual VQ.
// Block = 512 threads (16 waves), one head x 256 tokens; wave owns 16 tokens.
// argmax(qn . cn) == argmax(residual . cn): the q-norm is a positive scalar.
// ---------------------------------------------------------------------------
__global__ __launch_bounds__(NTHREADS) void rvq_main_kernel(
    const float* __restrict__ x, const float* __restrict__ cb,
    const _Float16* __restrict__ cn, float* __restrict__ out,
    float* __restrict__ idx_out, int* __restrict__ counts) {
  extern __shared__ char smem[];
  _Float16* sCB = (_Float16*)smem;                                  // [M][D] f16
  float* sRes = (float*)(smem + LDS_CB_BYTES);                      // [256][D] f32
  int* sIdx = (int*)(smem + LDS_CB_BYTES + LDS_RES_BYTES);          // [16][16]

  const int head = blockIdx.x & 7;
  const int tokenBase = (blockIdx.x >> 3) * TOK_PER_BLOCK;
  const int tid = threadIdx.x;
  const int w = tid >> 5;
  const int lane = tid & 31;

  // Stage normalized f16 codebook for this head into LDS (128 KB).
  {
    const uint4* src = (const uint4*)(cn + (size_t)head * MM * DD);
    uint4* dst = (uint4*)sCB;
    const int n16 = (MM * DD * 2) / 16;
    for (int i = tid; i < n16; i += NTHREADS) dst[i] = src[i];
  }
  // Stage residual = x slice for this head into LDS (64 KB).
  for (int i = tid; i < TOK_PER_BLOCK * DD / 4; i += NTHREADS) {
    const int row = i >> 4;
    const int col = (i & 15) * 4;
    float4 v = *(const float4*)(x + (size_t)(tokenBase + row) * FF + head * DD + col);
    *(float4*)(sRes + row * DD + col) = v;
  }
  __syncthreads();

  const int mrow = lane & 15;          // A: token row in tile; B/C: column
  const int hiA = (lane >> 4) * 8;     // accumulator (C/D) row offset per lane half
  const int hiB = (lane >> 4) * 16;    // B-fragment K sub-offset per lane half
  float* myRes = sRes + (w * 16 + mrow) * DD;
  const _Float16* bbase = sCB + mrow * DD + hiB;

  for (int r = 0; r < RR; ++r) {
    // Build f16 A fragments from the f32 residual (K = 0..63 as two frags).
    v16h a0, a1;
#pragma unroll
    for (int e = 0; e < 8; ++e) {
      a0[e]     = (_Float16)myRes[hiB + e];
      a0[e + 8] = (_Float16)myRes[8 + hiB + e];
      a1[e]     = (_Float16)myRes[32 + hiB + e];
      a1[e + 8] = (_Float16)myRes[40 + hiB + e];
    }

    float best[8];
    int bidx[8];
#pragma unroll
    for (int j = 0; j < 8; ++j) { best[j] = -3.4e38f; bidx[j] = 0; }

    // Sweep 1024 codewords: two 16-wide tiles per iteration, double-buffered
    // B fragments so DS latency overlaps the WMMAs, and two independent
    // accumulators so argmax VALU fills WMMA->VALU hazard slots.
    v16h b0A, b1A, b0B, b1B;
    loadB(bbase, b0A, b1A);
    loadB(bbase + 16 * DD, b0B, b1B);
    for (int m0 = 0; m0 < MM; m0 += 32) {
      const int mn = (m0 + 32) & (MM - 1);   // wraps to 0 on last iteration
      v16h n0A, n1A, n0B, n1B;
      loadB(bbase + mn * DD, n0A, n1A);
      loadB(bbase + (mn + 16) * DD, n0B, n1B);

      v8f c0 = {};
      c0 = __builtin_amdgcn_wmma_f32_16x16x32_f16(false, a0, false, b0A,
                                                  (short)0, c0, false, false);
      c0 = __builtin_amdgcn_wmma_f32_16x16x32_f16(false, a1, false, b1A,
                                                  (short)0, c0, false, false);
      v8f c1 = {};
      c1 = __builtin_amdgcn_wmma_f32_16x16x32_f16(false, a0, false, b0B,
                                                  (short)0, c1, false, false);
      c1 = __builtin_amdgcn_wmma_f32_16x16x32_f16(false, a1, false, b1B,
                                                  (short)0, c1, false, false);

      const int idxA = m0 + mrow;
      const int idxB = m0 + 16 + mrow;
#pragma unroll
      for (int j = 0; j < 8; ++j) {
        bool tA = c0[j] > best[j];
        best[j] = tA ? c0[j] : best[j];
        bidx[j] = tA ? idxA : bidx[j];
      }
#pragma unroll
      for (int j = 0; j < 8; ++j) {
        bool tB = c1[j] > best[j];
        best[j] = tB ? c1[j] : best[j];
        bidx[j] = tB ? idxB : bidx[j];
      }
      b0A = n0A; b1A = n1A; b0B = n0B; b1B = n1B;
    }

    // Branch-free argmax reduction across the 16 lanes of each half
    // (tie -> smaller idx, matching jnp.argmax first-occurrence).
#pragma unroll
    for (int off = 1; off < 16; off <<= 1) {
#pragma unroll
      for (int j = 0; j < 8; ++j) {
        float ob = __shfl_xor(best[j], off, 32);
        int oi = __shfl_xor(bidx[j], off, 32);
        bool take = (ob > best[j]) | ((ob == best[j]) & (oi < bidx[j]));
        best[j] = take ? ob : best[j];
        bidx[j] = take ? oi : bidx[j];
      }
    }

    // Lanes 0 and 16 hold the winners for rows hiA..hiA+7 of the wave tile.
    if (mrow == 0) {
#pragma unroll
      for (int j = 0; j < 8; ++j) {
        const int row = hiA + j;                 // 0..15 within wave tile
        const int t = tokenBase + w * 16 + row;  // global token
        sIdx[w * 16 + row] = bidx[j];
        const int bb = t >> 11;                  // / 2048
        const int nn = t & 2047;
        // indices layout: [b][h][n*R + r]
        idx_out[((size_t)(bb * HH + head)) * (NN * RR) + nn * RR + r] =
            (float)bidx[j];
        atomicAdd(&counts[(r * HH + head) * MM + bidx[j]], 1);
      }
    }

    // residual -= codebooks[head, idx] (UNnormalized f32, from L2).
    // Wave-local rows only -> no block barrier needed inside the r loop.
    for (int j2 = 0; j2 < 16; ++j2) {
      const int idx = sIdx[w * 16 + j2];
      const float* q = cb + ((size_t)(head * MM + idx)) * DD + lane * 2;
      float* rr = sRes + (w * 16 + j2) * DD + lane * 2;
      rr[0] -= q[0];
      rr[1] -= q[1];
    }
  }
  __syncthreads();

  // out = sum of quants = x - residual_final
  for (int i = tid; i < TOK_PER_BLOCK * DD / 4; i += NTHREADS) {
    const int row = i >> 4;
    const int col = (i & 15) * 4;
    const size_t g = (size_t)(tokenBase + row) * FF + head * DD + col;
    float4 xv = *(const float4*)(x + g);
    float4 rv = *(const float4*)(sRes + row * DD + col);
    float4 ov = make_float4(xv.x - rv.x, xv.y - rv.y, xv.z - rv.z, xv.w - rv.w);
    *(float4*)(out + g) = ov;
  }
}

// ---------------------------------------------------------------------------
// Kernel 2: perplexity = exp(-sum p*log(p+eps)) per (r,h); output in (h r).
// ---------------------------------------------------------------------------
__global__ void perp_kernel(const int* __restrict__ counts,
                            float* __restrict__ perp_out) {
  __shared__ float red[8];
  const int rh = blockIdx.x;           // r*H + h
  const int r = rh / HH;
  const int h = rh % HH;
  const int* c = counts + rh * MM;
  float s = 0.0f;
  for (int i = threadIdx.x; i < MM; i += 256) {
    float p = (float)c[i] * (1.0f / (float)TT);
    s += p * logf(p + 1e-10f);
  }
#pragma unroll
  for (int off = 16; off >= 1; off >>= 1) s += __shfl_xor(s, off, 32);
  const int w = threadIdx.x >> 5;
  const int lane = threadIdx.x & 31;
  if (lane == 0) red[w] = s;
  __syncthreads();
  if (threadIdx.x == 0) {
    float t = 0.0f;
#pragma unroll
    for (int i = 0; i < 8; ++i) t += red[i];
    perp_out[h * RR + r] = expf(-t);
  }
}

// ---------------------------------------------------------------------------
extern "C" void kernel_launch(void* const* d_in, const int* in_sizes, int n_in,
                              void* d_out, int out_size, void* d_ws,
                              size_t ws_size, hipStream_t stream) {
  const float* x = (const float*)d_in[0];          // [8,2048,512]
  const float* cb = (const float*)d_in[1];         // [8,1024,64]
  float* out = (float*)d_out;                      // [8,2048,512]
  float* idx_out = out + (size_t)BB * NN * FF;     // [8,8,2048*4] as float
  float* perp_out = idx_out + (size_t)BB * HH * NN * RR;  // [32]

  _Float16* cn = (_Float16*)d_ws;                               // 1 MB
  int* counts = (int*)((char*)d_ws + (size_t)HH * MM * DD * 2); // 128 KB

  (void)hipFuncSetAttribute((const void*)rvq_main_kernel,
                            hipFuncAttributeMaxDynamicSharedMemorySize,
                            LDS_TOTAL);

  cb_norm_kernel<<<(HH * MM) / 8, 256, 0, stream>>>(cb, cn);
  zero_kernel<<<(RR * HH * MM + 255) / 256, 256, 0, stream>>>(counts,
                                                              RR * HH * MM);
  rvq_main_kernel<<<(TT / TOK_PER_BLOCK) * HH, NTHREADS, LDS_TOTAL, stream>>>(
      x, cb, cn, out, idx_out, counts);
  perp_kernel<<<RR * HH, 256, 0, stream>>>(counts, perp_out);
}
